// ModifiedLSTM_31207232373044
// MI455X (gfx1250) — matrix operand order repaired and checked by reading, not compiled, not run on hardware
//
#include <hip/hip_runtime.h>
#include <hip/hip_bf16.h>
#include <math.h>

// Problem constants
#define Bq   64
#define Tq   2048
#define Iq   128
#define Hq   256
#define G4H  1024
#define Cq   64
#define TCq  128              // timesteps per chunk (xg chunk = 32MB, L2-resident)
#define NREC_WG 16

typedef __attribute__((ext_vector_type(16))) _Float16 v16h;
typedef __attribute__((ext_vector_type(8)))  _Float16 v8h;
typedef __attribute__((ext_vector_type(8)))  float    v8f;

// ---------------------------------------------------------------- helpers

__device__ __forceinline__ float sigf(float x) {
  return 1.0f / (1.0f + __expf(-x));
}
__device__ __forceinline__ float tanh_fast(float x) {
  // tanh(x) = 2*sigmoid(2x) - 1  (one v_exp_f32)
  return 2.0f / (1.0f + __expf(-2.0f * x)) - 1.0f;
}

// A-frag: 16x32 f16, row-major source, lane(0-15)=row M, lane>=16 same rows K+8
__device__ __forceinline__ v16h load_a(const _Float16* rowbase, int k0, int half) {
  const v8h lo = *(const v8h*)(rowbase + k0 + 8 * half);        // K = k0+8h .. +7
  const v8h hi = *(const v8h*)(rowbase + k0 + 16 + 8 * half);   // K = k0+16+8h ..
  v16h a;
#pragma unroll
  for (int i = 0; i < 8; i++) { a[i] = lo[i]; a[i + 8] = hi[i]; }
  return a;
}

// B-frag: 32x16 f16. Column n of B = row n of row-major W[N][K]; lanes 0-15 hold
// K=k0..k0+7, k0+16..k0+23 for col n=lane, lanes 16-31 hold K=k0+8..+15, k0+24..+31.
__device__ __forceinline__ v16h load_b(const _Float16* W, int ldk, int col, int k0, int half) {
  const _Float16* p = W + (size_t)col * ldk + k0 + 8 * half;
  const v8h lo = *(const v8h*)p;
  const v8h hi = *(const v8h*)(p + 16);
  v16h b;
#pragma unroll
  for (int i = 0; i < 8; i++) { b[i] = lo[i]; b[i + 8] = hi[i]; }
  return b;
}

// device-wide barrier for the persistent recurrence kernel (16 co-resident WGs)
__device__ __forceinline__ void grid_sync(int* cnt, volatile int* gen, int nwg) {
  __threadfence();
  __syncthreads();
  if (threadIdx.x == 0) {
    int g = *gen;
    if (atomicAdd(cnt, 1) == nwg - 1) {
      atomicExch(cnt, 0);
      __threadfence();
      *gen = g + 1;
    } else {
      while (*gen == g) { __builtin_amdgcn_s_sleep(2); }
    }
  }
  __syncthreads();
  __threadfence();   // acquire: next step's h loads must see remote stores
}

// ---------------------------------------------------------------- prep kernels

__global__ void cvt_f16_kernel(const float* __restrict__ in, _Float16* __restrict__ out, int n) {
  int i = blockIdx.x * blockDim.x + threadIdx.x;
  int stride = gridDim.x * blockDim.x;
  for (; i < n; i += stride) out[i] = (_Float16)in[i];
}

__global__ void bias_sum_kernel(const float* __restrict__ a, const float* __restrict__ b,
                                float* __restrict__ out, int n) {
  int i = blockIdx.x * blockDim.x + threadIdx.x;
  if (i < n) out[i] = a[i] + b[i];
}

__global__ void init_state_kernel(_Float16* h_buf, float* c_buf, int* bar) {
  int i = blockIdx.x * blockDim.x + threadIdx.x;
  int stride = gridDim.x * blockDim.x;
  for (int k = i; k < 2 * 64 * Hq; k += stride) h_buf[k] = (_Float16)0.0f;
  for (int k = i; k < 64 * Hq; k += stride) c_buf[k] = 0.0f;
  if (i < 2) bar[i] = 0;
}

// ---------------------------------------------------------------- xg = x @ Wih^T + (bih+bhh)
// One chunk of TCq timesteps: M = B*TCq = 8192 rows, N = 1024, K = 128 or 256.
// 8 waves/block, each wave computes one 16(M) x 64(N) strip (A-frag reuse x4).

__global__ void __launch_bounds__(256) xg_gemm_kernel(
    const _Float16* __restrict__ X,    // [B, T, K] f16
    const _Float16* __restrict__ Wih,  // [1024, K] f16
    const float*    __restrict__ bsum, // [1024]
    float*          __restrict__ xg,   // [B*TCq, 1024]
    int K, int t0)
{
  const int lane = threadIdx.x & 31;
  const int wid  = threadIdx.x >> 5;
  const int q    = blockIdx.x * 8 + wid;   // 0..8191
  const int m16  = q >> 4;                 // 512 M-tiles
  const int n4   = q & 15;                 // 16 N-strips of 64
  const int row  = lane & 15;
  const int half = lane >> 4;

  // output row -> (batch, tc) -> source row pointer (per lane)
  const int r_out = m16 * 16 + row;
  const int b  = r_out >> 7;               // TCq = 128
  const int tc = r_out & (TCq - 1);
  const _Float16* arow = X + (size_t)(b * Tq + t0 + tc) * K;

  v8f acc[4];
#pragma unroll
  for (int nt = 0; nt < 4; nt++)
#pragma unroll
    for (int r = 0; r < 8; r++) acc[nt][r] = 0.0f;

  const int colbase = n4 * 64;
  for (int k0 = 0; k0 < K; k0 += 32) {
    const v16h a = load_a(arow, k0, half);
#pragma unroll
    for (int nt = 0; nt < 4; nt++) {
      const v16h bf = load_b(Wih, K, colbase + nt * 16 + row, k0, half);
      acc[nt] = __builtin_amdgcn_wmma_f32_16x16x32_f16(
          false, a, false, bf, (short)0, acc[nt], false, false);
    }
  }
#pragma unroll
  for (int nt = 0; nt < 4; nt++) {
    const int col = colbase + nt * 16 + row;
    const float bv = bsum[col];
#pragma unroll
    for (int r = 0; r < 8; r++) {
      const int orow = m16 * 16 + r + 8 * half;
      xg[(size_t)orow * G4H + col] = acc[nt][r] + bv;
    }
  }
}

// ---------------------------------------------------------------- persistent LSTM recurrence
// 16 WGs x 512 threads (16 waves). WG wg owns hidden slice j0=wg*16 -> 64 gate
// columns (i,f,g,o x 16). Wave (mt = wid&3, gate = wid>>2) computes one 16x16
// tile: batch rows mt*16.., gate cols gate*256+j0... Whh frags live in VGPRs.
// h ping-pongs in a global f16 buffer (L2), one grid barrier per step.

__global__ void __launch_bounds__(512) lstm_rec_kernel(
    const float*    __restrict__ xg,     // [B*TCq, 1024]
    const _Float16* __restrict__ Whh,    // [1024, 256] f16
    const float*    __restrict__ mask,   // [B, T]
    _Float16*       __restrict__ h_buf,  // [2][64][256] ping-pong
    float*          __restrict__ c_buf,  // [64][256]
    _Float16*       __restrict__ h_all,  // [B, T, 256] layer output
    int t0, int* bar_cnt, volatile int* bar_gen)
{
  __shared__ float ldsg[4 * 64 * 16];    // [gate][batch][hidden] staging, 16KB

  const int tid  = threadIdx.x;
  const int lane = tid & 31;
  const int wid  = tid >> 5;     // 0..15
  const int mt   = wid & 3;      // batch tile
  const int gate = wid >> 2;     // 0..3 = i,f,g,o
  const int j0   = blockIdx.x * 16;
  const int row  = lane & 15;
  const int half = lane >> 4;

  // preload Whh fragments for this wave's gate-column tile (8 x v16h, stays in regs)
  v16h bf[8];
#pragma unroll
  for (int k = 0; k < 8; k++)
    bf[k] = load_b(Whh, Hq, gate * Hq + j0 + row, k * 32, half);

  // c-state in registers: this thread owns elements e={tid, tid+512} of the
  // WG's 64x16 slice; e -> (batch=e>>4, hidden=j0+(e&15))
  const int b0 = tid >> 4,         jj = tid & 15;
  const int b1 = (tid + 512) >> 4;
  float c0 = c_buf[b0 * Hq + j0 + jj];
  float c1 = c_buf[b1 * Hq + j0 + jj];

  for (int tc = 0; tc < TCq; ++tc) {
    const int t = t0 + tc;
    const _Float16* hrd = h_buf + (size_t)(t & 1) * 64 * Hq;
    _Float16*       hwr = h_buf + (size_t)((t + 1) & 1) * 64 * Hq;

    // init accumulator with precomputed xg tile
    v8f acc;
#pragma unroll
    for (int r = 0; r < 8; r++) {
      const int brow = mt * 16 + r + 8 * half;
      acc[r] = xg[(size_t)(brow * TCq + tc) * G4H + gate * Hq + j0 + row];
    }
    if (tc + 1 < TCq)  // pull next step's xg tile toward L0 (global_prefetch_b8)
      __builtin_prefetch(&xg[(size_t)((mt * 16 + row) * TCq + tc + 1) * G4H + gate * Hq + j0], 0, 0);

    // acc += h_old @ Whh_slice^T   (8 x v_wmma_f32_16x16x32_f16)
#pragma unroll
    for (int k = 0; k < 8; k++) {
      const v16h a = load_a(hrd + (size_t)(mt * 16 + row) * Hq, k * 32, half);
      acc = __builtin_amdgcn_wmma_f32_16x16x32_f16(
          false, a, false, bf[k], (short)0, acc, false, false);
    }

    // stage gates to LDS for cross-wave i/f/g/o recombination
#pragma unroll
    for (int r = 0; r < 8; r++) {
      const int brow = mt * 16 + r + 8 * half;
      ldsg[(gate * 64 + brow) * 16 + row] = acc[r];
    }
    __syncthreads();

    // elementwise LSTM update, 2 cells per thread
    {
      const float i0 = ldsg[(0 * 64 + b0) * 16 + jj];
      const float f0 = ldsg[(1 * 64 + b0) * 16 + jj];
      const float g0 = ldsg[(2 * 64 + b0) * 16 + jj];
      const float o0 = ldsg[(3 * 64 + b0) * 16 + jj];
      const float i1 = ldsg[(0 * 64 + b1) * 16 + jj];
      const float f1 = ldsg[(1 * 64 + b1) * 16 + jj];
      const float g1 = ldsg[(2 * 64 + b1) * 16 + jj];
      const float o1 = ldsg[(3 * 64 + b1) * 16 + jj];
      const float m0 = mask[b0 * Tq + t];
      const float m1 = mask[b1 * Tq + t];

      c0 = sigf(f0) * c0 + sigf(i0) * tanh_fast(g0);
      float h0 = sigf(o0) * tanh_fast(c0);
      h0 *= m0; c0 *= m0;
      c1 = sigf(f1) * c1 + sigf(i1) * tanh_fast(g1);
      float h1 = sigf(o1) * tanh_fast(c1);
      h1 *= m1; c1 *= m1;

      hwr[b0 * Hq + j0 + jj] = (_Float16)h0;
      hwr[b1 * Hq + j0 + jj] = (_Float16)h1;
      h_all[((size_t)b0 * Tq + t) * Hq + j0 + jj] = (_Float16)h0;
      h_all[((size_t)b1 * Tq + t) * Hq + j0 + jj] = (_Float16)h1;
    }
    grid_sync(bar_cnt, bar_gen, NREC_WG);   // publish h for all WGs
  }
  c_buf[b0 * Hq + j0 + jj] = c0;
  c_buf[b1 * Hq + j0 + jj] = c1;
}

// ---------------------------------------------------------------- LN + ReLU (layer0 -> layer1 input)

__global__ void __launch_bounds__(256) ln_relu_kernel(
    const _Float16* __restrict__ hin, const float* __restrict__ g,
    const float* __restrict__ be, _Float16* __restrict__ out)
{
  const int lane = threadIdx.x & 31, wid = threadIdx.x >> 5;
  const size_t rowi = (size_t)blockIdx.x * 8 + wid;   // < B*T
  const _Float16* p = hin + rowi * Hq;
  float v[8], s = 0.f, s2 = 0.f;
#pragma unroll
  for (int k = 0; k < 8; k++) {
    v[k] = (float)p[lane + 32 * k];
    s += v[k]; s2 += v[k] * v[k];
  }
#pragma unroll
  for (int o = 16; o > 0; o >>= 1) {
    s  += __shfl_xor(s,  o, 32);
    s2 += __shfl_xor(s2, o, 32);
  }
  const float mu = s * (1.0f / Hq);
  const float var = s2 * (1.0f / Hq) - mu * mu;
  const float rs = rsqrtf(var + 1e-5f);
  _Float16* q = out + rowi * Hq;
#pragma unroll
  for (int k = 0; k < 8; k++) {
    const int j = lane + 32 * k;
    const float y = (v[k] - mu) * rs * g[j] + be[j];
    q[j] = (_Float16)(y > 0.f ? y : 0.f);
  }
}

// ---------------------------------------------------------------- head: LN+ReLU on t=T-1, then FC

__global__ void __launch_bounds__(256) head_kernel(
    const _Float16* __restrict__ h_all, const float* __restrict__ g,
    const float* __restrict__ be, const float* __restrict__ fcW,
    const float* __restrict__ fcb, float* __restrict__ out)
{
  __shared__ float sh[Hq];
  __shared__ float red[Hq];
  const int b = blockIdx.x, j = threadIdx.x;
  const float v = (float)h_all[((size_t)b * Tq + (Tq - 1)) * Hq + j];
  red[j] = v; __syncthreads();
  for (int s = 128; s > 0; s >>= 1) { if (j < s) red[j] += red[j + s]; __syncthreads(); }
  const float mu = red[0] * (1.0f / Hq); __syncthreads();
  const float d = v - mu;
  red[j] = d * d; __syncthreads();
  for (int s = 128; s > 0; s >>= 1) { if (j < s) red[j] += red[j + s]; __syncthreads(); }
  const float var = red[0] * (1.0f / Hq);
  const float y = d * rsqrtf(var + 1e-5f) * g[j] + be[j];
  sh[j] = y > 0.f ? y : 0.f;
  __syncthreads();
  if (j < Cq) {
    float a = fcb[j];
    for (int k = 0; k < Hq; k++) a += sh[k] * fcW[j * Hq + k];
    out[b * Cq + j] = a;
  }
}

// ---------------------------------------------------------------- launch

extern "C" void kernel_launch(void* const* d_in, const int* in_sizes, int n_in,
                              void* d_out, int out_size, void* d_ws, size_t ws_size,
                              hipStream_t stream) {
  (void)in_sizes; (void)n_in; (void)out_size; (void)ws_size;
  const float* seq  = (const float*)d_in[0];
  const float* mask = (const float*)d_in[1];
  const float* Wih[2] = { (const float*)d_in[2], (const float*)d_in[8]  };
  const float* Whh[2] = { (const float*)d_in[3], (const float*)d_in[9]  };
  const float* bih[2] = { (const float*)d_in[4], (const float*)d_in[10] };
  const float* bhh[2] = { (const float*)d_in[5], (const float*)d_in[11] };
  const float* lng[2] = { (const float*)d_in[6], (const float*)d_in[12] };
  const float* lnb[2] = { (const float*)d_in[7], (const float*)d_in[13] };
  const float* fcW = (const float*)d_in[14];
  const float* fcb = (const float*)d_in[15];
  float* out = (float*)d_out;

  char* ws = (char*)d_ws; size_t off = 0;
  auto take = [&](size_t bytes) -> void* {
    void* p = ws + off; off += (bytes + 255) & ~(size_t)255; return p;
  };
  _Float16* x0h    = (_Float16*)take((size_t)Bq * Tq * Iq * 2);   // 32 MB
  _Float16* x1h    = (_Float16*)take((size_t)Bq * Tq * Hq * 2);   // 64 MB
  _Float16* h_allh = (_Float16*)take((size_t)Bq * Tq * Hq * 2);   // 64 MB
  float*    xg     = (float*)   take((size_t)Bq * TCq * G4H * 4); // 32 MB
  _Float16* wih0h  = (_Float16*)take((size_t)G4H * Iq * 2);
  _Float16* whh0h  = (_Float16*)take((size_t)G4H * Hq * 2);
  _Float16* wih1h  = (_Float16*)take((size_t)G4H * Hq * 2);
  _Float16* whh1h  = (_Float16*)take((size_t)G4H * Hq * 2);
  float*    bsum0  = (float*)   take((size_t)G4H * 4);
  float*    bsum1  = (float*)   take((size_t)G4H * 4);
  _Float16* h_buf  = (_Float16*)take((size_t)2 * 64 * Hq * 2);
  float*    c_buf  = (float*)   take((size_t)64 * Hq * 4);
  int*      bar    = (int*)     take(256);

  // precision/layout prep: fp32 -> f16 operands, fused bias
  cvt_f16_kernel<<<2048, 256, 0, stream>>>(seq, x0h, Bq * Tq * Iq);
  cvt_f16_kernel<<<512, 256, 0, stream>>>(Wih[0], wih0h, G4H * Iq);
  cvt_f16_kernel<<<512, 256, 0, stream>>>(Whh[0], whh0h, G4H * Hq);
  cvt_f16_kernel<<<512, 256, 0, stream>>>(Wih[1], wih1h, G4H * Hq);
  cvt_f16_kernel<<<512, 256, 0, stream>>>(Whh[1], whh1h, G4H * Hq);
  bias_sum_kernel<<<4, 256, 0, stream>>>(bih[0], bhh[0], bsum0, G4H);
  bias_sum_kernel<<<4, 256, 0, stream>>>(bih[1], bhh[1], bsum1, G4H);

  for (int layer = 0; layer < 2; ++layer) {
    const _Float16* xin  = layer ? x1h   : x0h;
    const _Float16* wihh = layer ? wih1h : wih0h;
    const _Float16* whhh = layer ? whh1h : whh0h;
    const float*    bs   = layer ? bsum1 : bsum0;
    const int K = layer ? Hq : Iq;

    init_state_kernel<<<128, 256, 0, stream>>>(h_buf, c_buf, bar);
    for (int ch = 0; ch < Tq / TCq; ++ch) {
      const int t0 = ch * TCq;
      xg_gemm_kernel<<<1024, 256, 0, stream>>>(xin, wihh, bs, xg, K, t0);
      lstm_rec_kernel<<<NREC_WG, 512, 0, stream>>>(xg, whhh, mask, h_buf, c_buf,
                                                   h_allh, t0, bar, bar + 1);
    }
    if (layer == 0)
      ln_relu_kernel<<<Bq * Tq / 8, 256, 0, stream>>>(h_allh, lng[0], lnb[0], x1h);
  }
  head_kernel<<<Bq, 256, 0, stream>>>(h_allh, lng[1], lnb[1], fcW, fcb, out);
}
